// MultiHeadAttentionQuantum_65481071403110
// MI455X (gfx1250) — matrix operand order repaired and checked by reading, not compiled, not run on hardware
//
#include <hip/hip_runtime.h>
#include <cstdint>
#include <cstddef>

#define NH    16      // heads
#define DK    8       // d_k
#define EE    128     // embed = NH*DK
#define TOK   16      // tokens per tile
#define SEQ   8192    // S
#define SDIV  512     // S / TOK
#define BPAD  136     // padded bf16 row stride (272 B -> 4-bank rotate, conflict-free)
#define TPB   8       // tiles per (persistent) workgroup
#define NBLK  1024    // 8192 tiles / TPB

typedef __attribute__((ext_vector_type(16))) __bf16       v16bf;
typedef __attribute__((ext_vector_type(8)))  float        v8f;
typedef __attribute__((ext_vector_type(4)))  unsigned int v4u;

union Frag { v16bf bf; v4u q[2]; };

__global__ __launch_bounds__(256) void qattn_fused_kernel(
    const float* __restrict__ x, const float* __restrict__ theta,
    const float* __restrict__ Wc, const float* __restrict__ bc,
    float* __restrict__ out)
{
  __shared__ float  qkv[TOK][EE];          // 8 KB  : cos(x+theta) tile
  __shared__ __bf16 Ahi[NH][BPAD];         // attention-out tile, hi part
  __shared__ __bf16 Alo[NH][BPAD];         // lo part
  __shared__ __bf16 Bhi[EE][BPAD];         // 34 KB : Wc[j][e] hi  (WMMA B, [n][k])
  __shared__ __bf16 Blo[EE][BPAD];         // 34 KB : Wc[j][e] lo
  __shared__ float  thetaS[DK];

  const int tid = threadIdx.x;

  // ---- Phase 0 (once per persistent WG): Wc -> split bf16 in LDS.
  // B[k=e][n=j] = Wc[j][e]; storing Wc row-major as [j][e] gives
  // contiguous-K rows for the B fragment.
  for (int i = tid; i < EE * EE; i += 256) {
    const int j = i >> 7, e = i & 127;
    const float w  = Wc[i];
    const __bf16 hi = (__bf16)w;
    Bhi[j][e] = hi;
    Blo[j][e] = (__bf16)(w - (float)hi);
  }
  if (tid < DK) thetaS[tid] = theta[tid];

  // Per-lane constants for the WMMA phase (hoisted out of the tile loop)
  const int lane = tid & 31;
  const int wave = tid >> 5;
  const int n0   = wave * 16;
  const int mn   = lane & 15;              // A row (head) == D col within tile
  const int hl   = (lane >> 4) & 1;        // lane half
  const float bias = bc[n0 + mn];

  __syncthreads();

  for (int it = 0; it < TPB; ++it) {
    const int gt = blockIdx.x * TPB + it;  // global tile id
    const int b  = gt >> 9;                // batch
    const int r  = gt & 511;               // 16-token group within sequence

    // ---- Phase 2: qkv = cos(x + theta), float4 (b128) loads
    const float4* xt = (const float4*)(x + ((size_t)b * SEQ + (size_t)r * TOK) * EE);
    #pragma unroll
    for (int k = 0; k < (TOK * EE / 4) / 256; ++k) {
      const int f4 = tid + k * 256;        // float4 index within tile (0..511)
      const float4 v = xt[f4];
      const int t  = f4 >> 5;              // 32 float4 per token row
      const int e0 = (f4 * 4) & 127;
      const int d0 = e0 & 7;               // 0 or 4
      qkv[t][e0 + 0] = __cosf(v.x + thetaS[d0 + 0]);
      qkv[t][e0 + 1] = __cosf(v.y + thetaS[d0 + 1]);
      qkv[t][e0 + 2] = __cosf(v.z + thetaS[d0 + 2]);
      qkv[t][e0 + 3] = __cosf(v.w + thetaS[d0 + 3]);
    }

    // Prefetch next tile's x (the only HBM-resident stream) while we compute.
    if (it + 1 < TPB && lane == 0) {
      const float* xn = x + ((size_t)(gt + 1) >> 9) * SEQ * EE
                          + ((size_t)(gt + 1) & 511) * TOK * EE;
      // 8 waves x 1 lane -> 8 prefetches spanning the 8 KB tile
      __builtin_prefetch(xn + wave * (TOK * EE / 8), 0, 1);
    }
    __syncthreads();

    // ---- Phase 3: attention, one (token, head) softmax row per thread.
    {
      const int t = tid >> 4;              // token 0..15
      const int h = tid & 15;              // head  0..15
      float q[DK];
      #pragma unroll
      for (int d = 0; d < DK; ++d) q[d] = qkv[t][h * DK + d];

      float sc[NH];
      float mx = -3.0e38f;
      #pragma unroll
      for (int g = 0; g < NH; ++g) {
        float s = 0.f;
        #pragma unroll
        for (int d = 0; d < DK; ++d) s = fmaf(q[d], qkv[t][g * DK + d], s);
        s *= 0.35355339059327373f;         // 1/sqrt(d_k)
        sc[g] = s;
        mx = fmaxf(mx, s);
      }
      float sum = 0.f;
      #pragma unroll
      for (int g = 0; g < NH; ++g) { sc[g] = __expf(sc[g] - mx); sum += sc[g]; }
      const float inv = 1.0f / sum;

      float o[DK];
      #pragma unroll
      for (int d = 0; d < DK; ++d) o[d] = 0.f;
      #pragma unroll
      for (int g = 0; g < NH; ++g) {
        const float p = sc[g] * inv;
        #pragma unroll
        for (int d = 0; d < DK; ++d) o[d] = fmaf(p, qkv[t][g * DK + d], o[d]);
      }
      // Scrambled reshape: A row = head h, col = t*8 + d (split bf16 hi/lo)
      #pragma unroll
      for (int d = 0; d < DK; ++d) {
        const float  v  = o[d];
        const __bf16 hi = (__bf16)v;
        Ahi[h][t * DK + d] = hi;
        Alo[h][t * DK + d] = (__bf16)(v - (float)hi);
      }
    }
    __syncthreads();

    // ---- Phase 4: O = A x Wc^T via split-bf16 WMMA (3 products), fp32 acc.
    // 8 waves, one 16-wide N tile each; K = 128 -> 4 steps of 16x16x32.
    v8f acc = {};
    #pragma unroll
    for (int kt = 0; kt < 4; ++kt) {
      // A fragment: lane half selects K+8; elements 0..7 at ka, 8..15 at ka+16
      const int ka = kt * 32 + hl * 8;
      Frag aH, aL, bH, bL;
      aH.q[0] = *(const v4u*)&Ahi[mn][ka];
      aH.q[1] = *(const v4u*)&Ahi[mn][ka + 16];
      aL.q[0] = *(const v4u*)&Alo[mn][ka];
      aL.q[1] = *(const v4u*)&Alo[mn][ka + 16];
      // B fragment: lane half selects K+8; elements 0..7 at kb, 8..15 at kb+16
      const int kb = kt * 32 + hl * 8;
      bH.q[0] = *(const v4u*)&Bhi[n0 + mn][kb];
      bH.q[1] = *(const v4u*)&Bhi[n0 + mn][kb + 16];
      bL.q[0] = *(const v4u*)&Blo[n0 + mn][kb];
      bL.q[1] = *(const v4u*)&Blo[n0 + mn][kb + 16];

      acc = __builtin_amdgcn_wmma_f32_16x16x32_bf16(false, aH.bf, false, bH.bf,
                                                    (short)0, acc, false, false);
      acc = __builtin_amdgcn_wmma_f32_16x16x32_bf16(false, aH.bf, false, bL.bf,
                                                    (short)0, acc, false, false);
      acc = __builtin_amdgcn_wmma_f32_16x16x32_bf16(false, aL.bf, false, bH.bf,
                                                    (short)0, acc, false, false);
    }

    // ---- Phase 5: scatter D (+bias). Output row for head hr is hr*512 + r.
    #pragma unroll
    for (int g = 0; g < 8; ++g) {
      const int    hr  = g + hl * 8;                 // D row = head
      const size_t row = (size_t)hr * SDIV + r;      // scrambled token index
      out[((size_t)b * SEQ + row) * EE + (n0 + mn)] = acc[g] + bias;
    }
  }
}

extern "C" void kernel_launch(void* const* d_in, const int* in_sizes, int n_in,
                              void* d_out, int out_size, void* d_ws, size_t ws_size,
                              hipStream_t stream) {
  (void)in_sizes; (void)n_in; (void)out_size; (void)d_ws; (void)ws_size;
  const float* x     = (const float*)d_in[0];
  const float* theta = (const float*)d_in[1];
  const float* Wc    = (const float*)d_in[2];
  const float* bc    = (const float*)d_in[3];
  float* out = (float*)d_out;

  dim3 grid(NBLK);        // 1024 persistent WGs x 8 tiles
  dim3 block(256);        // 8 wave32
  qattn_fused_kernel<<<grid, block, 0, stream>>>(x, theta, Wc, bc, out);
}
